// GPT_LLM_84095459655854
// MI455X (gfx1250) — matrix operand order repaired and checked by reading, not compiled, not run on hardware
//
#include <hip/hip_runtime.h>
#include <hip/hip_bf16.h>

// ---------------- GPT forward on gfx1250 (CDNA5, wave32, WMMA) --------------
// B=32 T=64 E=256 H=8 D=32 L=8 F=1024 V=50257.
// All GEMMs: v_wmma_f32_16x16x32_f16, operands pre-packed into the exact
// per-lane fragment layout so the GEMM hot loop is LDS-free and barrier-free.
// Wave tile 32x64: per k-step 12 coalesced global_load_b128 feed 8 WMMAs.

typedef __attribute__((ext_vector_type(16))) _Float16 v16h;
typedef __attribute__((ext_vector_type(8)))  float    v8f;

#define NB      32
#define NT      64
#define NE      256
#define NH      8
#define ND      32
#define NL      8
#define NF      1024
#define NV      50257
#define NVPAD   50304          // multiple of 128
#define NM      (NB * NT)      // 2048 token rows

// Fragment layouts (halves), one 16x32 (A) / 32x16 (B) tile = 512 halves:
//   A: lane = (m&15) + ((k>>4)&1)*16 ; slot = k&15
//   B: lane = (n&15) + ((k>>4)&1)*16 ; slot = k&15
//   element offset = frag*512 + lane*16 + slot
//   frag = tile_index * (K/32) + k_chunk   (K-contiguous streaming)

#define WMMA_F16(a, b, c) \
  __builtin_amdgcn_wmma_f32_16x16x32_f16(false, (a), false, (b), (short)0, (c), false, false)

// ---------------------------------------------------------------------------
// Fragment-packed GEMM: C[M,N] = act(A*B + bias).
// Block: 128 threads (4 waves), tile 64(M) x 128(N); wave tile 32x64
// (2 A-frags x 4 B-frags -> 8 independent WMMA accumulation chains).
// ---------------------------------------------------------------------------
__global__ __launch_bounds__(128) void gemm_frag_wmma(
    const _Float16* __restrict__ Ap,   // frag-packed A  [M x K]
    const _Float16* __restrict__ Bp,   // frag-packed B  [K x N]
    float* __restrict__ C, int ldc,
    int K, int Nreal,
    const float* __restrict__ bias, int relu)
{
  const int lane = threadIdx.x & 31;
  const int wave = threadIdx.x >> 5;  // 0..3
  const int mi = wave & 1;            // 32-row half of block tile
  const int ni = wave >> 1;           // 64-col half of block tile
  const int m0 = blockIdx.y * 64 + mi * 32;
  const int n0 = blockIdx.x * 128 + ni * 64;
  const int KC = K >> 5;              // k-chunks
  const size_t tileStride = (size_t)KC << 9;   // halves per 16-wide tile panel

  const _Float16* a0 = Ap + (size_t)(m0 >> 4) * tileStride + lane * 16;
  const _Float16* a1 = a0 + tileStride;
  const _Float16* b0 = Bp + (size_t)(n0 >> 4) * tileStride + lane * 16;
  const _Float16* b1 = b0 + tileStride;
  const _Float16* b2 = b1 + tileStride;
  const _Float16* b3 = b2 + tileStride;

  v8f acc[2][4] = {};

  for (int kc = 0; kc < KC; ++kc) {
    const size_t o = (size_t)kc << 9;
    v16h A0 = *(const v16h*)(a0 + o);
    v16h A1 = *(const v16h*)(a1 + o);
    v16h B0 = *(const v16h*)(b0 + o);
    v16h B1 = *(const v16h*)(b1 + o);
    v16h B2 = *(const v16h*)(b2 + o);
    v16h B3 = *(const v16h*)(b3 + o);
    acc[0][0] = WMMA_F16(A0, B0, acc[0][0]);
    acc[0][1] = WMMA_F16(A0, B1, acc[0][1]);
    acc[0][2] = WMMA_F16(A0, B2, acc[0][2]);
    acc[0][3] = WMMA_F16(A0, B3, acc[0][3]);
    acc[1][0] = WMMA_F16(A1, B0, acc[1][0]);
    acc[1][1] = WMMA_F16(A1, B1, acc[1][1]);
    acc[1][2] = WMMA_F16(A1, B2, acc[1][2]);
    acc[1][3] = WMMA_F16(A1, B3, acc[1][3]);
  }

  const int half16 = lane >> 4;
  const int nlane  = lane & 15;
#pragma unroll
  for (int nn = 0; nn < 4; ++nn) {
    const int col = n0 + nn * 16 + nlane;
    const bool ok = col < Nreal;
    float bv = (bias && ok) ? bias[col] : 0.0f;
#pragma unroll
    for (int mm = 0; mm < 2; ++mm) {
      const int rowB = m0 + mm * 16 + half16 * 8;
#pragma unroll
      for (int r = 0; r < 8; ++r) {
        float v = acc[mm][nn][r] + bv;
        if (relu) v = fmaxf(v, 0.0f);
        if (ok) C[(size_t)(rowB + r) * ldc + col] = v;
      }
    }
  }
}

// ---------------------------------------------------------------------------
// LDS fragment loads for the attention kernel (data must transit LDS anyway
// for the softmax).  Same per-lane layouts as above.
// ---------------------------------------------------------------------------
__device__ inline v16h lds_frag_a(const _Float16* As, int stride,
                                  int mBase, int kBase, int lane) {
  const int half16 = lane >> 4;
  const int m      = lane & 15;
  const _Float16* row = As + (size_t)(mBase + m) * stride;
  const int k0 = kBase + half16 * 16;
  const int k1 = kBase + half16 * 16 + 8;
  v16h a;
#pragma unroll
  for (int i = 0; i < 8; ++i) { a[i] = row[k0 + i]; a[8 + i] = row[k1 + i]; }
  return a;
}

__device__ inline v16h lds_frag_b(const _Float16* Bs, int stride,
                                  int nBase, int kBase, int lane) {
  const int half16 = lane >> 4;
  const int n      = lane & 15;
  const _Float16* row = Bs + (size_t)(nBase + n) * stride;  // [n][k]
  const int k0 = kBase + half16 * 16;
  v16h b;
#pragma unroll
  for (int i = 0; i < 16; ++i) b[i] = row[k0 + i];
  return b;
}

// ---------------------------------------------------------------------------
// Fused attention, one workgroup (128 thr, 4 waves) per (b,h).
// qkv: [2048][768], cols [0,256)=Q [256,512)=K [512,768)=V; head slice h*32.
// ---------------------------------------------------------------------------
__global__ __launch_bounds__(128) void attention_wmma(
    const float* __restrict__ qkv, float* __restrict__ att)
{
  const int b = blockIdx.x >> 3;
  const int h = blockIdx.x & 7;

  __shared__ _Float16 Qs[64][40];   // A layout [t][d]
  __shared__ _Float16 Ks[64][40];   // B-trans layout [s][d]
  __shared__ _Float16 Vts[32][72];  // B-trans layout [d][s]
  __shared__ float    Sf[64][66];
  __shared__ _Float16 Ps[64][72];   // A layout [t][s]

  const int tid  = threadIdx.x;
  const int lane = tid & 31;
  const int wave = tid >> 5;
  const float* base = qkv + (size_t)b * 64 * 768;

  // Stage Q,K: 64 rows x 32 cols as 8-float groups (vector loads/stores).
#pragma unroll
  for (int i = 0; i < 2; ++i) {
    int e = tid + i * 128;            // 0..255
    int r = e >> 2, g = (e & 3) * 8;
    const float* qsrc = base + r * 768 + h * 32 + g;
    const float* ksrc = base + r * 768 + 256 + h * 32 + g;
#pragma unroll
    for (int j = 0; j < 8; ++j) Qs[r][g + j] = (_Float16)qsrc[j];
#pragma unroll
    for (int j = 0; j < 8; ++j) Ks[r][g + j] = (_Float16)ksrc[j];
  }
  // Stage V transposed: 4-float groups, scatter across Vts rows.
#pragma unroll
  for (int i = 0; i < 4; ++i) {
    int e = tid + i * 128;            // 0..511
    int t = e >> 3, g = (e & 7) * 4;
    const float* vsrc = base + t * 768 + 512 + h * 32 + g;
#pragma unroll
    for (int j = 0; j < 4; ++j) Vts[g + j][t] = (_Float16)vsrc[j];
  }
  __syncthreads();

  const int half16 = lane >> 4, nlane = lane & 15;
  const float scale = 0.17677669529663687f;  // 1/sqrt(32)

  // S = Q K^T : 16 rows/wave, N=64, K=32.
  v16h aq = lds_frag_a(&Qs[0][0], 40, wave * 16, 0, lane);
#pragma unroll
  for (int nt = 0; nt < 4; ++nt) {
    v16h bk = lds_frag_b(&Ks[0][0], 40, nt * 16, 0, lane);
    v8f acc = {};
    acc = WMMA_F16(aq, bk, acc);
#pragma unroll
    for (int r = 0; r < 8; ++r) {
      int row = wave * 16 + half16 * 8 + r;
      int col = nt * 16 + nlane;
      Sf[row][col] = (col <= row) ? acc[r] * scale : -__builtin_inff();
    }
  }
  __syncthreads();

  // Causal softmax, thread t owns row t.
  if (tid < 64) {
    float mx = -__builtin_inff();
    for (int s = 0; s < 64; ++s) mx = fmaxf(mx, Sf[tid][s]);
    float sum = 0.0f;
    for (int s = 0; s < 64; ++s) sum += __expf(Sf[tid][s] - mx);
    float inv = 1.0f / sum;
    for (int s = 0; s < 64; ++s)
      Ps[tid][s] = (_Float16)(__expf(Sf[tid][s] - mx) * inv);
  }
  __syncthreads();

  // att = P V : M=64, N=32, K=64.
  v8f o0 = {}, o1 = {};
#pragma unroll
  for (int kc = 0; kc < 64; kc += 32) {
    v16h ap  = lds_frag_a(&Ps[0][0], 72, wave * 16, kc, lane);
    v16h bv0 = lds_frag_b(&Vts[0][0], 72, 0,  kc, lane);
    v16h bv1 = lds_frag_b(&Vts[0][0], 72, 16, kc, lane);
    o0 = WMMA_F16(ap, bv0, o0);
    o1 = WMMA_F16(ap, bv1, o1);
  }
  float* outp = att + (size_t)b * 64 * 256 + h * 32;
#pragma unroll
  for (int r = 0; r < 8; ++r) {
    int row = wave * 16 + half16 * 8 + r;
    outp[(size_t)row * 256 + nlane]      = o0[r];
    outp[(size_t)row * 256 + 16 + nlane] = o1[r];
  }
}

// ---------------------------------------------------------------------------
// x = LayerNorm(x + y) * g + b  (y may be null).  One block per row, E=256.
// ---------------------------------------------------------------------------
__global__ __launch_bounds__(256) void add_layernorm(
    float* __restrict__ x, const float* __restrict__ y,
    const float* __restrict__ g, const float* __restrict__ bta)
{
  __shared__ float red[256];
  const int row = blockIdx.x;
  const int c   = threadIdx.x;
  float v = x[(size_t)row * NE + c];
  if (y) v += y[(size_t)row * NE + c];

  red[c] = v; __syncthreads();
  for (int s = 128; s > 0; s >>= 1) { if (c < s) red[c] += red[c + s]; __syncthreads(); }
  float mean = red[0] * (1.0f / NE);
  __syncthreads();

  float d = v - mean;
  red[c] = d * d; __syncthreads();
  for (int s = 128; s > 0; s >>= 1) { if (c < s) red[c] += red[c + s]; __syncthreads(); }
  float var = red[0] * (1.0f / NE);

  x[(size_t)row * NE + c] = d * rsqrtf(var + 1e-5f) * g[c] + bta[c];
}

// ---------------------------------------------------------------------------
// x[row] = tok_emb[ip[row]] + pos_emb[row % T]
// ---------------------------------------------------------------------------
__global__ __launch_bounds__(256) void embed_k(
    const int* __restrict__ ip, const float* __restrict__ tok,
    const float* __restrict__ pos, float* __restrict__ x)
{
  const int row = blockIdx.x;
  const int c   = threadIdx.x;
  const int t   = row & (NT - 1);
  const int tk  = ip[row];
  x[(size_t)row * NE + c] = tok[(size_t)tk * NE + c] + pos[(size_t)t * NE + c];
}

// ---------------------------------------------------------------------------
// Activation pack: fp32 [M][K] row-major -> f16 A-fragment layout.
// Each thread handles 8 contiguous k (one aligned 16-byte output run).
// ---------------------------------------------------------------------------
__global__ __launch_bounds__(256) void pack_a_frag(
    const float* __restrict__ A, _Float16* __restrict__ Ap,
    int Kdim, int total8)
{
  int idx = blockIdx.x * 256 + threadIdx.x;
  if (idx >= total8) return;
  const int kg = Kdim >> 3;
  const int m  = idx / kg;
  const int k  = (idx - m * kg) << 3;
  const int KC = Kdim >> 5;
  const int frag = (m >> 4) * KC + (k >> 5);
  const int lane = (m & 15) + (((k >> 4) & 1) << 4);
  const int slot = ((k >> 3) & 1) << 3;
  const float* src = A + (size_t)m * Kdim + k;
  _Float16* dst = Ap + ((size_t)frag << 9) + lane * 16 + slot;
#pragma unroll
  for (int i = 0; i < 8; ++i) dst[i] = (_Float16)src[i];
}

// ---------------------------------------------------------------------------
// Weight pack: fp32 row-major [Kdim][ldn] -> f16 B-fragment layout,
// columns >= Nsrc zero-filled.  One thread per output half (coalesced writes).
// ---------------------------------------------------------------------------
__global__ __launch_bounds__(256) void pack_b_frag(
    const float* __restrict__ W, _Float16* __restrict__ Bp,
    int Kdim, int ldn, int Nsrc, int total)
{
  int idx = blockIdx.x * 256 + threadIdx.x;
  if (idx >= total) return;
  const int frag = idx >> 9;
  const int r    = idx & 511;
  const int lane = r >> 4, slot = r & 15;
  const int KC = Kdim >> 5;
  const int nt = frag / KC, kc = frag - nt * KC;
  const int n = nt * 16 + (lane & 15);
  const int k = (kc << 5) + ((lane >> 4) << 4) + slot;
  float v = (n < Nsrc) ? W[(size_t)k * ldn + n] : 0.0f;
  Bp[idx] = (_Float16)v;
}

// Fused QKV weight pack: gather Wq/Wk/Wv (L,H,E,D) into per-layer B-fragment
// layout of the [E, 3E] fused matrix.
__global__ __launch_bounds__(256) void pack_qkv_frag(
    const float* __restrict__ Wq, const float* __restrict__ Wk,
    const float* __restrict__ Wv, _Float16* __restrict__ out, int total)
{
  int idx = blockIdx.x * 256 + threadIdx.x;
  if (idx >= total) return;
  const int per = NE * 3 * NE;               // halves per layer
  const int l  = idx / per;
  const int r0 = idx - l * per;
  const int frag = r0 >> 9;
  const int r    = r0 & 511;
  const int lane = r >> 4, slot = r & 15;
  const int KC = NE / 32;                    // 8
  const int nt = frag / KC, kc = frag - nt * KC;
  const int n = nt * 16 + (lane & 15);       // 0..767
  const int k = (kc << 5) + ((lane >> 4) << 4) + slot;  // E index
  const int sel = n >> 8;                    // 0=Q 1=K 2=V
  const int jj  = n & 255;
  const int hh  = jj >> 5, d = jj & 31;
  const float* src = (sel == 0) ? Wq : (sel == 1) ? Wk : Wv;
  out[idx] = (_Float16)src[(((size_t)l * NH + hh) * NE + k) * ND + d];
}

// ---------------------------------------------------------------------------
// Host launcher
// ---------------------------------------------------------------------------
extern "C" void kernel_launch(void* const* d_in, const int* in_sizes, int n_in,
                              void* d_out, int out_size, void* d_ws, size_t ws_size,
                              hipStream_t stream) {
  (void)in_sizes; (void)n_in; (void)out_size; (void)ws_size;

  const int*   ip    = (const int*)  d_in[0];
  const float* tok   = (const float*)d_in[1];
  const float* pos   = (const float*)d_in[2];
  const float* Wq    = (const float*)d_in[3];
  const float* Wk    = (const float*)d_in[4];
  const float* Wv    = (const float*)d_in[5];
  const float* Wo    = (const float*)d_in[6];
  const float* bo    = (const float*)d_in[7];
  const float* ln1g  = (const float*)d_in[8];
  const float* ln1b  = (const float*)d_in[9];
  const float* ln2g  = (const float*)d_in[10];
  const float* ln2b  = (const float*)d_in[11];
  const float* W1    = (const float*)d_in[12];
  const float* b1    = (const float*)d_in[13];
  const float* W2    = (const float*)d_in[14];
  const float* b2    = (const float*)d_in[15];
  const float* lnfg  = (const float*)d_in[16];
  const float* lnfb  = (const float*)d_in[17];
  const float* Wlm   = (const float*)d_in[18];
  const float* blm   = (const float*)d_in[19];
  float* out = (float*)d_out;

  // Workspace carve-up
  char* ws = (char*)d_ws;
  size_t off = 0;
  auto take = [&](size_t bytes) { char* p = ws + off; off += (bytes + 255) & ~(size_t)255; return p; };
  float*    x      = (float*)   take((size_t)NM * NE * 4);
  float*    qkv    = (float*)   take((size_t)NM * 3 * NE * 4);
  float*    attb   = (float*)   take((size_t)NM * NE * 4);
  float*    h1     = (float*)   take((size_t)NM * NF * 4);
  float*    tmp    = (float*)   take((size_t)NM * NE * 4);
  _Float16* ap256  = (_Float16*)take((size_t)NM * NE * 2);   // packed A, K=256
  _Float16* ap1024 = (_Float16*)take((size_t)NM * NF * 2);   // packed A, K=1024
  _Float16* Wqkv_p = (_Float16*)take((size_t)NL * NE * 3 * NE * 2);
  _Float16* Wo_p   = (_Float16*)take((size_t)NL * NE * NE * 2);
  _Float16* W1_p   = (_Float16*)take((size_t)NL * NE * NF * 2);
  _Float16* W2_p   = (_Float16*)take((size_t)NL * NF * NE * 2);
  _Float16* Wlm_p  = (_Float16*)take((size_t)NE * NVPAD * 2);

  // ---- weight packing (once per launch) ----
  {
    int tq = NL * NE * 3 * NE;
    pack_qkv_frag<<<(tq + 255) / 256, 256, 0, stream>>>(Wq, Wk, Wv, Wqkv_p, tq);
    for (int l = 0; l < NL; ++l) {
      pack_b_frag<<<(NE * NE + 255) / 256, 256, 0, stream>>>(
          Wo + (size_t)l * NE * NE, Wo_p + (size_t)l * NE * NE, NE, NE, NE, NE * NE);
      pack_b_frag<<<(NE * NF + 255) / 256, 256, 0, stream>>>(
          W1 + (size_t)l * NE * NF, W1_p + (size_t)l * NE * NF, NE, NF, NF, NE * NF);
      pack_b_frag<<<(NF * NE + 255) / 256, 256, 0, stream>>>(
          W2 + (size_t)l * NF * NE, W2_p + (size_t)l * NF * NE, NF, NE, NE, NF * NE);
    }
    int tlm = NE * NVPAD;
    pack_b_frag<<<(tlm + 255) / 256, 256, 0, stream>>>(Wlm, Wlm_p, NE, NV, NV, tlm);
  }

  const int t8_256  = NM * (NE >> 3);   // pack_a work items, K=256
  const int t8_1024 = NM * (NF >> 3);   // pack_a work items, K=1024

  // ---- embedding ----
  embed_k<<<NM, 256, 0, stream>>>(ip, tok, pos, x);
  pack_a_frag<<<(t8_256 + 255) / 256, 256, 0, stream>>>(x, ap256, NE, t8_256);

  // ---- transformer blocks ----
  for (int l = 0; l < NL; ++l) {
    // QKV: [2048,256] x [256,768]
    gemm_frag_wmma<<<dim3(3 * NE / 128, NM / 64), 128, 0, stream>>>(
        ap256, Wqkv_p + (size_t)l * NE * 3 * NE, qkv, 3 * NE, NE, 3 * NE, nullptr, 0);
    // fused attention
    attention_wmma<<<NB * NH, 128, 0, stream>>>(qkv, attb);
    // output projection + bo
    pack_a_frag<<<(t8_256 + 255) / 256, 256, 0, stream>>>(attb, ap256, NE, t8_256);
    gemm_frag_wmma<<<dim3(NE / 128, NM / 64), 128, 0, stream>>>(
        ap256, Wo_p + (size_t)l * NE * NE, tmp, NE, NE, NE, bo + (size_t)l * NE, 0);
    // x = LN1(x + attn_out)
    add_layernorm<<<NM, 256, 0, stream>>>(x, tmp, ln1g + (size_t)l * NE, ln1b + (size_t)l * NE);
    pack_a_frag<<<(t8_256 + 255) / 256, 256, 0, stream>>>(x, ap256, NE, t8_256);
    // FFN up + ReLU
    gemm_frag_wmma<<<dim3(NF / 128, NM / 64), 128, 0, stream>>>(
        ap256, W1_p + (size_t)l * NE * NF, h1, NF, NE, NF, b1 + (size_t)l * NF, 1);
    // FFN down
    pack_a_frag<<<(t8_1024 + 255) / 256, 256, 0, stream>>>(h1, ap1024, NF, t8_1024);
    gemm_frag_wmma<<<dim3(NE / 128, NM / 64), 128, 0, stream>>>(
        ap1024, W2_p + (size_t)l * NF * NE, tmp, NE, NF, NE, b2 + (size_t)l * NE, 0);
    // x = LN2(x + ffn_out)
    add_layernorm<<<NM, 256, 0, stream>>>(x, tmp, ln2g + (size_t)l * NE, ln2b + (size_t)l * NE);
    pack_a_frag<<<(t8_256 + 255) / 256, 256, 0, stream>>>(x, ap256, NE, t8_256);
  }

  // ---- final LN + LM head ----
  add_layernorm<<<NM, 256, 0, stream>>>(x, nullptr, lnfg, lnfb);
  pack_a_frag<<<(t8_256 + 255) / 256, 256, 0, stream>>>(x, ap256, NE, t8_256);
  gemm_frag_wmma<<<dim3(NVPAD / 128, NM / 64), 128, 0, stream>>>(
      ap256, Wlm_p, out, NV, NE, NV, blm, 0);
}
